// LSTM_ATTN_GNN_26362509263638
// MI455X (gfx1250) — matrix operand order repaired and checked by reading, not compiled, not run on hardware
//
#include <hip/hip_runtime.h>
#include <hip/hip_bf16.h>

typedef __attribute__((ext_vector_type(16))) __bf16 v16bf;
typedef __attribute__((ext_vector_type(8)))  float  v8f;
typedef __attribute__((ext_vector_type(4)))  unsigned int v4u;
typedef __attribute__((ext_vector_type(8)))  int v8i;
typedef __attribute__((ext_vector_type(4)))  int v4i;

#define TSTEPS 128
#define NBATCH 256
#define HUMN   20
#define HID    256

// ---------------------------------------------------------------------------
// WMMA helpers (bf16 16x16x32, f32 accumulate)
// ---------------------------------------------------------------------------
static __device__ inline v8f wmma_bf16(v16bf a, v16bf b, v8f c) {
  return __builtin_amdgcn_wmma_f32_16x16x32_bf16(
      /*neg_a=*/false, a, /*neg_b=*/false, b,
      /*c_mod=*/(short)0, c, /*reuse_a=*/false, /*reuse_b=*/false);
}

// A fragment: 16x32 bf16 tile, row-major, leading dim ld. base = &A[row0][kb].
//   lanes 0-15: M=lane, elems 0..15 = K 0..15
//   lanes16-31: M=lane-16, elems 0..15 = K 16..31
static __device__ inline v16bf frag_a(const __bf16* base, int ld, int lane) {
  int m = lane & 15, kh = lane >> 4;
  const __bf16* p = base + (size_t)m * ld + kh * 16;
  v16bf f;
#pragma unroll
  for (int i = 0; i < 8; ++i) f[i] = p[i];
#pragma unroll
  for (int i = 0; i < 8; ++i) f[8 + i] = p[8 + i];
  return f;
}

// B fragment from TRANSPOSED weights WT (N rows x Kp cols). base = &WT[col0][kb].
//   lanes 0-15: N=lane, K = kb+0..15 ; lanes 16-31: N=lane-16, K = kb+16..31
static __device__ inline v16bf frag_b(const __bf16* base, int ld, int lane) {
  int n = lane & 15, kh = lane >> 4;
  const __bf16* p = base + (size_t)n * ld + kh * 16;
  v16bf f;
#pragma unroll
  for (int i = 0; i < 16; ++i) f[i] = p[i];
  return f;
}

// ---------------------------------------------------------------------------
// Tensor Data Mover: 2D tile of `rows` rows x (row_qw * 8) bytes, contiguous
// rows, global -> LDS.  D# per CDNA5 ISA.
// ---------------------------------------------------------------------------
static __device__ inline void tdm_load_rows(const void* gsrc, unsigned lds_addr,
                                            unsigned rows, unsigned row_qw) {
  unsigned long long ga = (unsigned long long)(size_t)gsrc;
  v4u g0;
  g0[0] = 1u;                                   // count=1, user descriptor
  g0[1] = lds_addr;                             // LDS byte address
  g0[2] = (unsigned)ga;                         // global_addr[31:0]
  g0[3] = (unsigned)((ga >> 32) & 0x01FFFFFFull) | (2u << 30);  // [56:32] | type=2
  v8i g1;
  g1[0] = (int)(3u << 16);       // workgroup_mask=0, data_size=3 (8B)
  g1[1] = (int)(row_qw << 16);   // tensor_dim0[15:0]
  g1[2] = (int)(rows << 16);     // tensor_dim0[31:16]=0 | tensor_dim1[15:0]
  g1[3] = (int)(row_qw << 16);   // tensor_dim1[31:16]=0 | tile_dim0
  g1[4] = (int)rows;             // tile_dim1 | tile_dim2=0
  g1[5] = (int)row_qw;           // tensor_dim0_stride[31:0]
  g1[6] = 0;                     // stride hi | tensor_dim1_stride lo
  g1[7] = 0;
  v4i z4 = {0, 0, 0, 0};
#if __clang_major__ >= 23
  v8i z8 = {0, 0, 0, 0, 0, 0, 0, 0};
  __builtin_amdgcn_tensor_load_to_lds(g0, g1, z4, z4, z8, 0);
#else
  __builtin_amdgcn_tensor_load_to_lds(g0, g1, z4, z4, 0);
#endif
}

// ---------------------------------------------------------------------------
// Weight repack: f32 (K x N) -> bf16 transposed padded (Np x Kp)
// ---------------------------------------------------------------------------
__global__ void pack_weight_t(const float* __restrict__ W, __bf16* __restrict__ WT,
                              int K, int N, int Kp, int Np) {
  int idx = blockIdx.x * blockDim.x + threadIdx.x;
  if (idx >= Np * Kp) return;
  int n = idx / Kp, k = idx % Kp;
  float v = (n < N && k < K) ? W[(size_t)k * N + n] : 0.f;
  WT[idx] = (__bf16)v;
}

__global__ void cast_f32_bf16(const float* __restrict__ s, __bf16* __restrict__ d, int n) {
  int i = blockIdx.x * blockDim.x + threadIdx.x;
  if (i < n) d[i] = (__bf16)s[i];
}

__global__ void copy_f32_k(const float* __restrict__ s, float* __restrict__ d, int n) {
  int i = blockIdx.x * blockDim.x + threadIdx.x;
  if (i < n) d[i] = s[i];
}

// masked h for step t, cast to bf16 once (feeds all WMMA A-fragments)
__global__ void mask_cast_h(const float* __restrict__ h, const float* __restrict__ masks,
                            int t, __bf16* __restrict__ out) {
  int i = blockIdx.x * blockDim.x + threadIdx.x;
  if (i >= NBATCH * HID) return;
  int row = i >> 8;
  out[i] = (__bf16)(h[i] * masks[t * NBATCH + row]);
}

// ---------------------------------------------------------------------------
// Fused edge MLP (21->150->50, relu) + min over HUM + node_in concat/pack.
// One workgroup per (t,n); 20 human rows padded to 32, everything in LDS.
// ---------------------------------------------------------------------------
__global__ __launch_bounds__(256)
void edge_min_pack(const float* __restrict__ rn, const float* __restrict__ te,
                   const float* __restrict__ se,
                   const __bf16* __restrict__ eW1T, const float* __restrict__ eb1,
                   const __bf16* __restrict__ eW2T, const float* __restrict__ eb2,
                   __bf16* __restrict__ node_in /* 32768 x 64 */) {
  __shared__ __bf16 sEin[32][32];   // edge_in, cols 0..20 valid
  __shared__ __bf16 sH1[32][160];   // relu(layer1), cols 0..149 valid
  __shared__ float  sH2[32][64];    // relu(layer2), cols 0..49 valid
  const int r = blockIdx.x;         // row in [0, T*N)
  const int tid = threadIdx.x;

  for (int idx = tid; idx < 32 * 32; idx += 256) {
    int row = idx >> 5, c = idx & 31;
    float v = 0.f;
    if (row < HUMN) {
      if (c < 7)       v = rn[r * 7 + c];
      else if (c < 9)  v = te[r * 2 + (c - 7)];
      else if (c < 21) v = se[(size_t)r * HUMN * 12 + row * 12 + (c - 9)];
    }
    sEin[row][c] = (__bf16)v;
  }
  __syncthreads();

  const int wave = tid >> 5, lane = tid & 31;
  const int n = lane & 15, kh = lane >> 4;

  // layer 1: 2 M-tiles x 10 N-tiles, K=32 (one WMMA each)
  for (int tI = wave; tI < 20; tI += 8) {
    int mt = tI / 10, nt = tI % 10;
    v8f acc = {};
    v16bf a = frag_a(&sEin[mt * 16][0], 32, lane);
    v16bf b = frag_b(eW1T + (size_t)(nt * 16) * 32, 32, lane);
    acc = wmma_bf16(a, b, acc);
    int c = nt * 16 + n;
    float bs = (c < 150) ? eb1[c] : 0.f;
#pragma unroll
    for (int v = 0; v < 8; ++v) {
      float x = fmaxf(acc[v] + bs, 0.f);
      sH1[mt * 16 + kh * 8 + v][c] = (__bf16)x;
    }
  }
  __syncthreads();

  // layer 2: 2 M-tiles x 4 N-tiles, K=160 (5 WMMA each)
  {
    int mt = wave >> 2, nt = wave & 3;
    v8f acc = {};
    for (int kb = 0; kb < 160; kb += 32) {
      v16bf a = frag_a(&sH1[mt * 16][kb], 160, lane);
      v16bf b = frag_b(eW2T + (size_t)(nt * 16) * 160 + kb, 160, lane);
      acc = wmma_bf16(a, b, acc);
    }
    int c = nt * 16 + n;
    float bs = (c < 50) ? eb2[c] : 0.f;
#pragma unroll
    for (int v = 0; v < 8; ++v)
      sH2[mt * 16 + kh * 8 + v][c] = fmaxf(acc[v] + bs, 0.f);
  }
  __syncthreads();

  // min over the 20 valid rows + concat robot_obs -> node_in row (bf16, Kp=64)
  if (tid < 64) {
    float v;
    if (tid < 50) {
      float mn = sH2[0][tid];
#pragma unroll
      for (int h = 1; h < HUMN; ++h) mn = fminf(mn, sH2[h][tid]);
      v = mn;
    } else if (tid < 57) v = rn[r * 7 + (tid - 50)];
    else if (tid < 59)   v = te[r * 2 + (tid - 57)];
    else v = 0.f;
    node_in[(size_t)r * 64 + tid] = (__bf16)v;
  }
}

// ---------------------------------------------------------------------------
// Fused node MLP (59->150->64, relu), 16 rows per workgroup, 10 waves.
// ---------------------------------------------------------------------------
__global__ __launch_bounds__(320)
void node_mlp(const __bf16* __restrict__ node_in,
              const __bf16* __restrict__ nW1T, const float* __restrict__ nb1,
              const __bf16* __restrict__ nW2T, const float* __restrict__ nb2,
              __bf16* __restrict__ rnn_in /* 32768 x 64 */) {
  __shared__ __bf16 sH1[16][160];
  const int row0 = blockIdx.x * 16;
  const int wave = threadIdx.x >> 5, lane = threadIdx.x & 31;
  const int n = lane & 15, kh = lane >> 4;
  {
    int nt = wave;  // 0..9
    v8f acc = {};
    for (int kb = 0; kb < 64; kb += 32) {
      v16bf a = frag_a(node_in + (size_t)row0 * 64 + kb, 64, lane);
      v16bf b = frag_b(nW1T + (size_t)(nt * 16) * 64 + kb, 64, lane);
      acc = wmma_bf16(a, b, acc);
    }
    int c = nt * 16 + n;
    float bs = (c < 150) ? nb1[c] : 0.f;
#pragma unroll
    for (int v = 0; v < 8; ++v)
      sH1[kh * 8 + v][c] = (__bf16)fmaxf(acc[v] + bs, 0.f);
  }
  __syncthreads();
  if (wave < 4) {
    int nt = wave;
    v8f acc = {};
    for (int kb = 0; kb < 160; kb += 32) {
      v16bf a = frag_a(&sH1[0][kb], 160, lane);
      v16bf b = frag_b(nW2T + (size_t)(nt * 16) * 160 + kb, 160, lane);
      acc = wmma_bf16(a, b, acc);
    }
    int c = nt * 16 + n;  // < 64
    float bs = nb2[c];
#pragma unroll
    for (int v = 0; v < 8; ++v)
      rnn_in[(size_t)(row0 + kh * 8 + v) * 64 + c] = (__bf16)fmaxf(acc[v] + bs, 0.f);
  }
}

// ---------------------------------------------------------------------------
// GRU step. Grid (16,16) tiles over (batch 256, hid 256). 4 waves:
//   wave0: r = xr+hr (K=64+256), wave1: z = xz+hz, wave2: xn (K=64),
//   wave3: hn (K=256).  h_new = (1-z)*tanh(xn + r*hn) + z*(h*m)
// h-part B tiles (16x512B of gWhT) arrive via Tensor Data Mover while the
// x-part WMMAs run on globally-fed fragments; s_wait_tensorcnt gates the
// LDS-fed h-part K-loop.  hmb = bf16(h_prev * mask) precomputed once.
// ---------------------------------------------------------------------------
__global__ __launch_bounds__(128)
void gru_step(int t, const __bf16* __restrict__ rnn_in, const float* __restrict__ masks,
              const float* __restrict__ h_prev, const __bf16* __restrict__ hmb,
              const __bf16* __restrict__ gWxT, const __bf16* __restrict__ gWhT,
              const float* __restrict__ gbx, const float* __restrict__ gbh,
              float* __restrict__ h_next, float* __restrict__ outs) {
  __shared__ float  sG[4][16][16];
  __shared__ float  sHM[16][16];
  __shared__ __bf16 sBh[3][16 * HID];   // TDM-staged gWhT tiles (waves 0,1,3)
  const int bi = blockIdx.x, bj = blockIdx.y;
  const int wave = threadIdx.x >> 5, lane = threadIdx.x & 31;
  const int n = lane & 15, kh = lane >> 4;
  const int rowbase = bi * 16;
  const int gate = (wave < 3) ? wave : 2;
  const bool do_x = (wave <= 2);
  const bool do_h = (wave != 2);
  const int hslot = (wave == 3) ? 2 : wave;

  if (do_h) {
    tdm_load_rows(gWhT + (size_t)(gate * HID + bj * 16) * HID,
                  (unsigned)(size_t)&sBh[hslot][0], 16, HID / 4);
  }

  v8f acc = {};
  if (do_x) {  // overlaps with the TDM transfer
    for (int kb = 0; kb < 64; kb += 32) {
      v16bf a = frag_a(rnn_in + (size_t)(t * NBATCH + rowbase) * 64 + kb, 64, lane);
      v16bf b = frag_b(gWxT + (size_t)(gate * HID + bj * 16) * 64 + kb, 64, lane);
      acc = wmma_bf16(a, b, acc);
    }
  }
  if (do_h) {
    __builtin_amdgcn_s_wait_tensorcnt(0);
    for (int kb = 0; kb < HID; kb += 32) {
      v16bf a = frag_a(hmb + (size_t)rowbase * HID + kb, HID, lane);
      v16bf b = frag_b(&sBh[hslot][kb], HID, lane);
      acc = wmma_bf16(a, b, acc);
    }
  }
#pragma unroll
  for (int v = 0; v < 8; ++v) sG[wave][kh * 8 + v][n] = acc[v];

  for (int idx = threadIdx.x; idx < 256; idx += 128) {
    int rr = idx >> 4, cc = idx & 15;
    sHM[rr][cc] = h_prev[(size_t)(rowbase + rr) * HID + bj * 16 + cc] *
                  masks[t * NBATCH + rowbase + rr];
  }
  __syncthreads();

  for (int idx = threadIdx.x; idx < 256; idx += 128) {
    int rr = idx >> 4, cc = idx & 15;
    int ch = bj * 16 + cc;
    float pr = sG[0][rr][cc] + gbx[ch] + gbh[ch];
    float pz = sG[1][rr][cc] + gbx[HID + ch] + gbh[HID + ch];
    float xn = sG[2][rr][cc] + gbx[2 * HID + ch];
    float hn = sG[3][rr][cc] + gbh[2 * HID + ch];
    float r_ = 1.f / (1.f + expf(-pr));
    float z  = 1.f / (1.f + expf(-pz));
    float nn = tanhf(xn + r_ * hn);
    float hv = sHM[rr][cc];
    float hnew = (1.f - z) * nn + z * hv;
    h_next[(size_t)(rowbase + rr) * HID + ch] = hnew;
    outs[(size_t)(t * NBATCH + rowbase + rr) * HID + ch] = hnew;
  }
}

// ---------------------------------------------------------------------------
// WMMA GEMM (Kp == 256): out = act(A[M,256]bf16 @ W + bias), W as WT (N x 256).
// Grid (M/16, N/64), block 128 (4 waves = 4 N-tiles). act: 0 none, 2 tanh.
// Weight tiles staged LDS-side by the Tensor Data Mover.
// ---------------------------------------------------------------------------
__global__ __launch_bounds__(128)
void gemm_bf16(const __bf16* __restrict__ A, const __bf16* __restrict__ WT,
               const float* __restrict__ bias, int Kp, int Nld,
               float* outF, __bf16* outB, int act) {
  __shared__ __bf16 sB[4][16 * HID];
  const int row0 = blockIdx.x * 16;
  const int wave = threadIdx.x >> 5, lane = threadIdx.x & 31;
  const int nt = blockIdx.y * 4 + wave;
  const int n = lane & 15, kh = lane >> 4;

  tdm_load_rows(WT + (size_t)(nt * 16) * Kp, (unsigned)(size_t)&sB[wave][0],
                16, (unsigned)(Kp >> 2));

  v8f acc = {};
  __builtin_amdgcn_s_wait_tensorcnt(0);
  for (int kb = 0; kb < Kp; kb += 32) {
    if (kb + 32 < Kp) __builtin_prefetch(A + (size_t)row0 * Kp + kb + 32, 0, 0);
    v16bf a = frag_a(A + (size_t)row0 * Kp + kb, Kp, lane);
    v16bf b = frag_b(&sB[wave][kb], Kp, lane);
    acc = wmma_bf16(a, b, acc);
  }
  int c = nt * 16 + n;
  float bs = bias ? bias[c] : 0.f;
#pragma unroll
  for (int v = 0; v < 8; ++v) {
    float x = acc[v] + bs;
    if (act == 2) x = tanhf(x);
    int rr = row0 + kh * 8 + v;
    if (outF) outF[(size_t)rr * Nld + c] = x;
    if (outB) outB[(size_t)rr * Nld + c] = (__bf16)x;
  }
}

__global__ void critic_head(const __bf16* __restrict__ h, const float* __restrict__ clW,
                            const float* __restrict__ clb, float* __restrict__ out, int M) {
  int r = blockIdx.x * blockDim.x + threadIdx.x;
  if (r >= M) return;
  float s = clb[0];
  for (int k = 0; k < HID; ++k) s += (float)h[(size_t)r * HID + k] * clW[k];
  out[r] = s;
}

// ---------------------------------------------------------------------------
// Host orchestration
// ---------------------------------------------------------------------------
extern "C" void kernel_launch(void* const* d_in, const int* in_sizes, int n_in,
                              void* d_out, int out_size, void* d_ws, size_t ws_size,
                              hipStream_t stream) {
  (void)in_sizes; (void)n_in; (void)out_size; (void)ws_size;
  const float* rn   = (const float*)d_in[0];
  const float* te   = (const float*)d_in[1];
  const float* se   = (const float*)d_in[2];
  const float* rhx  = (const float*)d_in[3];
  const float* msk  = (const float*)d_in[4];
  const float* eW1  = (const float*)d_in[5];  const float* eb1 = (const float*)d_in[6];
  const float* eW2  = (const float*)d_in[7];  const float* eb2 = (const float*)d_in[8];
  const float* nW1  = (const float*)d_in[9];  const float* nb1 = (const float*)d_in[10];
  const float* nW2  = (const float*)d_in[11]; const float* nb2 = (const float*)d_in[12];
  const float* gWx  = (const float*)d_in[13]; const float* gWh = (const float*)d_in[14];
  const float* gbx  = (const float*)d_in[15]; const float* gbh = (const float*)d_in[16];
  const float* aW1  = (const float*)d_in[17]; const float* ab1 = (const float*)d_in[18];
  const float* aW2  = (const float*)d_in[19]; const float* ab2 = (const float*)d_in[20];
  const float* cW1  = (const float*)d_in[21]; const float* cb1 = (const float*)d_in[22];
  const float* cW2  = (const float*)d_in[23]; const float* cb2 = (const float*)d_in[24];
  const float* clW  = (const float*)d_in[25]; const float* clb = (const float*)d_in[26];
  float* out = (float*)d_out;

  const int M = TSTEPS * NBATCH;  // 32768
  size_t off = 0;
  auto alloc = [&](size_t bytes) -> void* {
    off = (off + 255) & ~(size_t)255;
    void* p = (char*)d_ws + off;
    off += bytes;
    return p;
  };
  __bf16* eW1T = (__bf16*)alloc(160 * 32 * 2);
  __bf16* eW2T = (__bf16*)alloc(64 * 160 * 2);
  __bf16* nW1T = (__bf16*)alloc(160 * 64 * 2);
  __bf16* nW2T = (__bf16*)alloc(64 * 160 * 2);
  __bf16* gWxT = (__bf16*)alloc(768 * 64 * 2);
  __bf16* gWhT = (__bf16*)alloc(768 * 256 * 2);
  __bf16* aW1T = (__bf16*)alloc(256 * 256 * 2);
  __bf16* aW2T = (__bf16*)alloc(256 * 256 * 2);
  __bf16* cW1T = (__bf16*)alloc(256 * 256 * 2);
  __bf16* cW2T = (__bf16*)alloc(256 * 256 * 2);
  __bf16* node_in = (__bf16*)alloc((size_t)M * 64 * 2);
  __bf16* rnn_in  = (__bf16*)alloc((size_t)M * 64 * 2);
  float*  hA   = (float*)alloc(NBATCH * HID * 4);
  float*  hB   = (float*)alloc(NBATCH * HID * 4);
  __bf16* hmb  = (__bf16*)alloc(NBATCH * HID * 2);
  float*  outs = (float*)alloc((size_t)M * HID * 4);
  __bf16* xb   = (__bf16*)alloc((size_t)M * HID * 2);
  __bf16* t1b  = (__bf16*)alloc((size_t)M * HID * 2);
  __bf16* t2b  = (__bf16*)alloc((size_t)M * HID * 2);

  auto packW = [&](const float* W, __bf16* WT, int K, int N, int Kp, int Np) {
    int tot = Np * Kp;
    hipLaunchKernelGGL(pack_weight_t, dim3((tot + 255) / 256), dim3(256), 0, stream,
                       W, WT, K, N, Kp, Np);
  };
  packW(eW1, eW1T, 21, 150, 32, 160);
  packW(eW2, eW2T, 150, 50, 160, 64);
  packW(nW1, nW1T, 59, 150, 64, 160);
  packW(nW2, nW2T, 150, 64, 160, 64);
  packW(gWx, gWxT, 64, 768, 64, 768);
  packW(gWh, gWhT, 256, 768, 256, 768);
  packW(aW1, aW1T, 256, 256, 256, 256);
  packW(aW2, aW2T, 256, 256, 256, 256);
  packW(cW1, cW1T, 256, 256, 256, 256);
  packW(cW2, cW2T, 256, 256, 256, 256);

  // edge MLP + min + concat -> node_in (bf16, Kp=64)
  hipLaunchKernelGGL(edge_min_pack, dim3(M), dim3(256), 0, stream,
                     rn, te, se, eW1T, eb1, eW2T, eb2, node_in);
  // node MLP -> rnn_in (bf16, Kp=64)
  hipLaunchKernelGGL(node_mlp, dim3(M / 16), dim3(320), 0, stream,
                     node_in, nW1T, nb1, nW2T, nb2, rnn_in);

  // GRU scan: 128 sequential steps, ping-pong h; masked bf16 h cast once/step
  for (int t = 0; t < TSTEPS; ++t) {
    const float* hin = (t == 0) ? rhx : (((t - 1) & 1) ? hA : hB);
    float* hout = (t & 1) ? hA : hB;
    hipLaunchKernelGGL(mask_cast_h, dim3((NBATCH * HID + 255) / 256), dim3(256), 0,
                       stream, hin, msk, t, hmb);
    hipLaunchKernelGGL(gru_step, dim3(NBATCH / 16, HID / 16), dim3(128), 0, stream,
                       t, rnn_in, msk, hin, hmb, gWxT, gWhT, gbx, gbh, hout, outs);
  }
  float* hlast = hA;  // hout at t=127 (odd)

  // pack GRU outputs to bf16 for actor/critic GEMMs
  {
    int n = M * HID;
    hipLaunchKernelGGL(cast_f32_bf16, dim3((n + 255) / 256), dim3(256), 0, stream,
                       outs, xb, n);
  }
  dim3 gg(M / 16, HID / 64), gb(128);
  // critic: tanh MLP then linear head
  hipLaunchKernelGGL(gemm_bf16, gg, gb, 0, stream, xb,  cW1T, cb1, HID, HID,
                     (float*)nullptr, t1b, 2);
  hipLaunchKernelGGL(gemm_bf16, gg, gb, 0, stream, t1b, cW2T, cb2, HID, HID,
                     (float*)nullptr, t2b, 2);
  hipLaunchKernelGGL(critic_head, dim3((M + 255) / 256), dim3(256), 0, stream,
                     t2b, clW, clb, out, M);
  // actor: tanh MLP, second layer straight into d_out
  hipLaunchKernelGGL(gemm_bf16, gg, gb, 0, stream, xb,  aW1T, ab1, HID, HID,
                     (float*)nullptr, t1b, 2);
  hipLaunchKernelGGL(gemm_bf16, gg, gb, 0, stream, t1b, aW2T, ab2, HID, HID,
                     out + M, (__bf16*)nullptr, 2);
  // h_new
  hipLaunchKernelGGL(copy_f32_k, dim3((NBATCH * HID + 255) / 256), dim3(256), 0, stream,
                     hlast, out + M + (size_t)M * HID, NBATCH * HID);
}
